// paren_mLSTM_25838523253441
// MI455X (gfx1250) — matrix operand order repaired and checked by reading, not compiled, not run on hardware
//
#include <hip/hip_runtime.h>

// ---------------------------------------------------------------------------
// Routed 2-cell LSTM, persistent single-workgroup kernel for gfx1250 (MI455X)
//   - bf16 WMMA (v_wmma_f32_16x16x32_bf16) for the gate GEMMs
//   - fp32 recurrence state (c in registers, h double-buffered in LDS as bf16)
//   - both cells computed densely; per-row select implements the routing
// ---------------------------------------------------------------------------

typedef __bf16 bf16_t;
typedef __bf16 v16bf __attribute__((ext_vector_type(16)));
typedef float  v8f   __attribute__((ext_vector_type(8)));

#define EMB_   256
#define HID_   256
#define SEQ_   512
#define BATCH_ 64
#define CELLS_ 2
#define KDIM_  (EMB_ + HID_)     // 512: contraction dim (x then h)
#define G4_    (4 * HID_)        // 1024 gate rows per cell
#define XA_W_  (EMB_ + 2 * HID_) // 768: emb | h_buf0 | h_buf1

__device__ __forceinline__ float sigf(float x) {
    return 1.0f / (1.0f + __expf(-x));
}
__device__ __forceinline__ float tanh_fast(float x) {
    // tanh(x) = 1 - 2/(e^{2x}+1)
    return 1.0f - 2.0f / (__expf(2.0f * x) + 1.0f);
}

// ---------------------------------------------------------------------------
// Prep: Wt[cell][g][k] (bf16, k contiguous: k<256 -> W_ih, else W_hh),
//       Bias[cell][g] = b_ih + b_hh (fp32)
// ---------------------------------------------------------------------------
__global__ void prep_weights(const float* __restrict__ Wih,
                             const float* __restrict__ Whh,
                             const float* __restrict__ bih,
                             const float* __restrict__ bhh,
                             bf16_t* __restrict__ Wt,
                             float* __restrict__ Bias) {
    const int n = CELLS_ * G4_ * KDIM_;
    for (int i = blockIdx.x * blockDim.x + threadIdx.x; i < n;
         i += gridDim.x * blockDim.x) {
        const int k = i % KDIM_;
        const int g = (i / KDIM_) % G4_;
        const int c = i / (KDIM_ * G4_);
        const float w = (k < EMB_)
                            ? Wih[((size_t)c * G4_ + g) * EMB_ + k]
                            : Whh[((size_t)c * G4_ + g) * HID_ + (k - EMB_)];
        Wt[i] = (bf16_t)w;
    }
    const int nb = CELLS_ * G4_;
    for (int i = blockIdx.x * blockDim.x + threadIdx.x; i < nb;
         i += gridDim.x * blockDim.x) {
        Bias[i] = bih[i] + bhh[i];
    }
}

// ---------------------------------------------------------------------------
// Persistent LSTM kernel: one workgroup, 512 threads = 16 wave32s.
// Each wave owns 4 (mtile, htile) pairs; per pair: 8 accum tiles
// (2 cells x 4 gates), K=512 contracted with 16 x wmma_f32_16x16x32_bf16.
// ---------------------------------------------------------------------------
__global__ __launch_bounds__(512) void lstm_persistent(
    const int* __restrict__ inp,       // [64, 512]
    const int* __restrict__ assign,    // [32000]
    const float* __restrict__ emb,     // [32000, 256]
    const bf16_t* __restrict__ Wt,     // [2, 1024, 512] bf16
    const float* __restrict__ Bias,    // [2, 1024]
    float* __restrict__ out)           // [64, 512] = h|c
{
    __shared__ bf16_t XA[BATCH_][XA_W_];  // 96 KB of the 320 KB WGP LDS
    __shared__ int acell[BATCH_];

    const int tid  = threadIdx.x;
    const int lane = tid & 31;
    const int wave = tid >> 5;
    const int lhi  = lane >> 4;        // 0: lanes 0-15, 1: lanes 16-31
    const int llo  = lane & 15;

    // zero the t=0 h read buffer (physical cols [256,512))
    for (int i = tid; i < BATCH_ * HID_; i += blockDim.x)
        XA[i >> 8][EMB_ + (i & 255)] = (bf16_t)0.0f;

    // per-wave tile-pair setup (4 pairs per wave; 64 pairs total)
    int mt[4], ht[4], col[4];
    float biasr[4][2][4];
    v8f cst[4];
#pragma unroll
    for (int j = 0; j < 4; ++j) {
        const int p = wave * 4 + j;
        mt[j]  = p >> 4;               // 0..3  (rows mt*16 .. mt*16+15)
        ht[j]  = p & 15;               // 0..15 (h cols ht*16 .. ht*16+15)
        col[j] = ht[j] * 16 + llo;     // this lane's output column (C/D: N=lane&15)
#pragma unroll
        for (int c = 0; c < 2; ++c)
#pragma unroll
            for (int q = 0; q < 4; ++q)
                biasr[j][c][q] = Bias[c * G4_ + q * HID_ + col[j]];
#pragma unroll
        for (int v = 0; v < 8; ++v) cst[j][v] = 0.0f;
    }

    for (int t = 0; t < SEQ_; ++t) {
        // ---- gather: routing table + embedding rows (bf16) into XA ----
        if (tid < BATCH_) acell[tid] = assign[inp[tid * SEQ_ + t]];
        for (int i = tid; i < BATCH_ * EMB_; i += blockDim.x) {
            const int b = i >> 8, e = i & 255;
            const int tok = inp[b * SEQ_ + t];
            XA[b][e] = (bf16_t)emb[(size_t)tok * EMB_ + e];
        }
        __syncthreads();

        const int hrd = (t & 1) * HID_;        // physical offset of h read buf
        const int hwr = ((t + 1) & 1) * HID_;  // physical offset of h write buf

#pragma unroll 1
        for (int j = 0; j < 4; ++j) {
            // accumulators seeded with per-column bias (splat over 8 VGPRs)
            v8f acc[2][4];
#pragma unroll
            for (int c = 0; c < 2; ++c)
#pragma unroll
                for (int q = 0; q < 4; ++q)
#pragma unroll
                    for (int v = 0; v < 8; ++v) acc[c][q][v] = biasr[j][c][q];

            const int arow = mt[j] * 16 + llo;  // A: M = lane&15
            const int akof = lhi * 8;           // A: lanes>=16 start at K+8
            const int bkof = lhi * 8;           // B: lanes>=16 start at K+8

#pragma unroll 2
            for (int k0 = 0; k0 < KDIM_; k0 += 32) {
                const int pk0 = (k0 >= EMB_) ? (k0 + hrd) : k0;  // remap h half
                union { v16bf v; uint4 u[2]; } A;
                A.u[0] = *reinterpret_cast<const uint4*>(&XA[arow][pk0 + akof]);
                A.u[1] = *reinterpret_cast<const uint4*>(&XA[arow][pk0 + akof + 16]);
#pragma unroll
                for (int c = 0; c < 2; ++c) {
#pragma unroll
                    for (int q = 0; q < 4; ++q) {
                        union { v16bf v; uint4 u[2]; } B;
                        const bf16_t* wp =
                            Wt + (((size_t)c * G4_ + q * HID_ + col[j]) * KDIM_ +
                                  k0 + bkof);
                        const uint4* bp = reinterpret_cast<const uint4*>(wp);
                        B.u[0] = bp[0];
                        B.u[1] = bp[2];
                        acc[c][q] = __builtin_amdgcn_wmma_f32_16x16x32_bf16(
                            false, A.v, false, B.v, (short)0, acc[c][q], false,
                            false);
                    }
                }
            }

            // ---- pointwise LSTM update, routed per row, all in registers ----
#pragma unroll
            for (int v = 0; v < 8; ++v) {
                const int row = mt[j] * 16 + v + (lhi << 3);  // C/D: M=v(+8)
                const int ce  = acell[row];
                const float iv = ce ? acc[1][0][v] : acc[0][0][v];
                const float fv = ce ? acc[1][1][v] : acc[0][1][v];
                const float gv = ce ? acc[1][2][v] : acc[0][2][v];
                const float ov = ce ? acc[1][3][v] : acc[0][3][v];
                const float cn = sigf(fv) * cst[j][v] + sigf(iv) * tanh_fast(gv);
                const float hn = sigf(ov) * tanh_fast(cn);
                cst[j][v] = cn;
                XA[row][EMB_ + hwr + col[j]] = (bf16_t)hn;
                if (t == SEQ_ - 1) {
                    out[row * (2 * HID_) + col[j]] = hn;
                    out[row * (2 * HID_) + HID_ + col[j]] = cn;
                }
            }
        }
        __syncthreads();
    }
}

// ---------------------------------------------------------------------------
extern "C" void kernel_launch(void* const* d_in, const int* in_sizes, int n_in,
                              void* d_out, int out_size, void* d_ws,
                              size_t ws_size, hipStream_t stream) {
    (void)in_sizes; (void)n_in; (void)out_size; (void)ws_size;
    const int*   inp    = (const int*)d_in[0];
    const int*   assign = (const int*)d_in[1];
    const float* emb    = (const float*)d_in[2];
    const float* Wih    = (const float*)d_in[3];
    const float* Whh    = (const float*)d_in[4];
    const float* bih    = (const float*)d_in[5];
    const float* bhh    = (const float*)d_in[6];
    float* out = (float*)d_out;

    bf16_t* Wt = (bf16_t*)d_ws;                       // 2*1024*512 bf16 = 2 MB
    float* Bias = (float*)((char*)d_ws +
                           (size_t)CELLS_ * G4_ * KDIM_ * sizeof(bf16_t));

    prep_weights<<<1024, 256, 0, stream>>>(Wih, Whh, bih, bhh, Wt, Bias);
    lstm_persistent<<<1, 512, 0, stream>>>(inp, assign, emb, Wt, Bias, out);
}
